// GATLayer_20512763806105
// MI455X (gfx1250) — matrix operand order repaired and checked by reading, not compiled, not run on hardware
//
#include <hip/hip_runtime.h>

#define NN 4096
#define DD 128
#define NEG_SLOPE 0.2f

typedef unsigned short ushort_t;
typedef __bf16       v16bf __attribute__((ext_vector_type(16)));
typedef float        v8f   __attribute__((ext_vector_type(8)));
typedef float        v4f   __attribute__((ext_vector_type(4)));
typedef int          v4i   __attribute__((ext_vector_type(4)));
typedef unsigned int v4u   __attribute__((ext_vector_type(4)));

__device__ __forceinline__ unsigned bf16_bits(float f) {
  unsigned u = __float_as_uint(f);
  u += 0x7FFFu + ((u >> 16) & 1u);   // round-to-nearest-even
  return u >> 16;
}
__device__ __forceinline__ unsigned pack2_bf16(float lo, float hi) {
  return bf16_bits(lo) | (bf16_bits(hi) << 16);
}

// ---------------------------------------------------------------------------
// Kernel A: h = x @ W (fp32), store hbT = bf16(h)^T [DD][NN], s1 = h@a1, s2 = h@a2
// 128 blocks x 256 threads, 32 rows per block. k-tiled through LDS (~21 KB).
// ---------------------------------------------------------------------------
__global__ __launch_bounds__(256) void gat_h_kernel(
    const float* __restrict__ x, const float* __restrict__ W,
    const float* __restrict__ attn_w,
    ushort_t* __restrict__ hbT, float* __restrict__ s1, float* __restrict__ s2) {
  __shared__ float Wl[32][DD];    // W k-chunk: [k in chunk][col]
  __shared__ float Xl[32][33];    // x chunk:   [row][k in chunk], padded
  __shared__ float s1l[32], s2l[32];

  const int tid  = threadIdx.x;
  const int row  = tid >> 3;            // 0..31 (local row)
  const int col0 = (tid & 7) * 16;      // 16 output cols per thread
  const int rg0  = blockIdx.x * 32;

  float acc[16] = {};
  for (int kt = 0; kt < 4; ++kt) {
    const int kbase = kt * 32;
    __syncthreads();
    for (int i = tid; i < 32 * DD; i += 256)
      Wl[i >> 7][i & 127] = W[(size_t)(kbase + (i >> 7)) * DD + (i & 127)];
    for (int i = tid; i < 32 * 32; i += 256)
      Xl[i >> 5][i & 31] = x[(size_t)(rg0 + (i >> 5)) * DD + kbase + (i & 31)];
    __syncthreads();
    for (int k = 0; k < 32; ++k) {
      const float xv = Xl[row][k];
#pragma unroll
      for (int c = 0; c < 16; ++c) acc[c] += xv * Wl[k][col0 + c];
    }
  }

  // partial attention scores
  float p1 = 0.f, p2 = 0.f;
#pragma unroll
  for (int c = 0; c < 16; ++c) {
    p1 += acc[c] * attn_w[col0 + c];
    p2 += acc[c] * attn_w[DD + col0 + c];
  }
  if (tid < 32) { s1l[tid] = 0.f; s2l[tid] = 0.f; }
  __syncthreads();
  atomicAdd(&s1l[row], p1);
  atomicAdd(&s2l[row], p2);

  // store h transposed as bf16 (column-major => contiguous K per output dim)
  const int rgl = rg0 + row;
#pragma unroll
  for (int c = 0; c < 16; ++c)
    hbT[(size_t)(col0 + c) * NN + rgl] = (ushort_t)bf16_bits(acc[c]);

  __syncthreads();
  if (tid < 32) { s1[rg0 + tid] = s1l[tid]; s2[rg0 + tid] = s2l[tid]; }
}

// ---------------------------------------------------------------------------
// Kernel A2: s2max = max_j s2[j]  (exact per-row softmax bound ingredient)
// ---------------------------------------------------------------------------
__global__ __launch_bounds__(256) void gat_s2max_kernel(
    const float* __restrict__ s2, float* __restrict__ s2max) {
  __shared__ float red[256];
  float m = -1e30f;
  for (int i = threadIdx.x; i < NN; i += 256) m = fmaxf(m, s2[i]);
  red[threadIdx.x] = m;
  __syncthreads();
  for (int s = 128; s > 0; s >>= 1) {
    if (threadIdx.x < s) red[threadIdx.x] = fmaxf(red[threadIdx.x], red[threadIdx.x + s]);
    __syncthreads();
  }
  if (threadIdx.x == 0) *s2max = red[0];
}

// ---------------------------------------------------------------------------
// Kernel B: fused streaming attention. 256 blocks (16 rows each) x 8 waves.
// Wave w handles columns [w*512, w*512+512) in 16 steps of K=32.
// Per step: compute P (16x32 bf16, built directly in WMMA A-layout),
// 8x v_wmma_f32_16x16x32_bf16 against hbT tiles, accumulate l and o.
// ---------------------------------------------------------------------------
__device__ __forceinline__ float p_edge(int a, float s2j, float dj, float gj,
                                        float s1r, float b, float eM) {
  float e = s1r + s2j + b;
  e = e > 0.f ? e : NEG_SLOPE * e;                 // leaky relu
  float w = __expf(-dj) * (fmaxf(__cosf(gj), 0.f) + 1e-6f);
  float p = __expf(e - eM) * w;                    // e <= eM always (monotone bound)
  return (a != 0) ? p : 0.f;                       // adj mask (-inf -> weight 0)
}

__global__ __launch_bounds__(256) void gat_attn_kernel(
    const int* __restrict__ adj, const float* __restrict__ dist,
    const float* __restrict__ ang, const ushort_t* __restrict__ hbT,
    const float* __restrict__ s1, const float* __restrict__ s2,
    const float* __restrict__ s2max, const float* __restrict__ attn_b,
    float* __restrict__ out) {
  __shared__ float o_lds[16 * DD];   // 8 KB combine buffer
  __shared__ float l_lds[16];
  __shared__ float s1_lds[16];

  const int tid  = threadIdx.x;
  const int lane = tid & 31;
  const int wave = tid >> 5;
  const int row0 = blockIdx.x * 16;

  if (tid < 16) { l_lds[tid] = 0.f; s1_lds[tid] = s1[row0 + tid]; }
  for (int i = tid; i < 16 * DD; i += 256) o_lds[i] = 0.f;
  __syncthreads();

  const int   r   = lane & 15;       // WMMA A-layout row
  const int   hi  = lane >> 4;       // lane half selects K sub-range
  const float bcv = attn_b[0];
  const float s1r = s1_lds[r];
  float eM = s1r + s2max[0] + bcv;
  eM = eM > 0.f ? eM : NEG_SLOPE * eM;             // exact row max of e

  const int rg = row0 + r;
  const int*   arow = adj  + (size_t)rg * NN;
  const float* drow = dist + (size_t)rg * NN;
  const float* grow = ang  + (size_t)rg * NN;

  v8f  c[8] = {};                    // o tile: 16 rows x 128 dims (8 C tiles)
  float lsum = 0.f;

  for (int st = 0; st < 16; ++st) {
    const int k0 = wave * 512 + st * 32;
    // A-layout j runs: lanes 0-15 own j in {0..7,16..23}, lanes 16-31 {8..15,24..31}
    const int jA = k0 + (hi ? 8 : 16 * 0);         // run1 (dwords 0..3)
    const int jB = k0 + (hi ? 24 : 16);            // run2 (dwords 4..7)

    // streamed one-shot data: non-temporal so hbT stays L2-resident
    v4i a0 = __builtin_nontemporal_load((const v4i*)(arow + jA));
    v4i a1 = __builtin_nontemporal_load((const v4i*)(arow + jA + 4));
    v4i a2 = __builtin_nontemporal_load((const v4i*)(arow + jB));
    v4i a3 = __builtin_nontemporal_load((const v4i*)(arow + jB + 4));
    v4f d0 = __builtin_nontemporal_load((const v4f*)(drow + jA));
    v4f d1 = __builtin_nontemporal_load((const v4f*)(drow + jA + 4));
    v4f d2 = __builtin_nontemporal_load((const v4f*)(drow + jB));
    v4f d3 = __builtin_nontemporal_load((const v4f*)(drow + jB + 4));
    v4f g0 = __builtin_nontemporal_load((const v4f*)(grow + jA));
    v4f g1 = __builtin_nontemporal_load((const v4f*)(grow + jA + 4));
    v4f g2 = __builtin_nontemporal_load((const v4f*)(grow + jB));
    v4f g3 = __builtin_nontemporal_load((const v4f*)(grow + jB + 4));
    v4f z0 = *(const v4f*)(s2 + jA);
    v4f z1 = *(const v4f*)(s2 + jA + 4);
    v4f z2 = *(const v4f*)(s2 + jB);
    v4f z3 = *(const v4f*)(s2 + jB + 4);

    float p[16];
#pragma unroll
    for (int i = 0; i < 4; ++i) {
      p[i]      = p_edge(a0[i], z0[i], d0[i], g0[i], s1r, bcv, eM);
      p[4 + i]  = p_edge(a1[i], z1[i], d1[i], g1[i], s1r, bcv, eM);
      p[8 + i]  = p_edge(a2[i], z2[i], d2[i], g2[i], s1r, bcv, eM);
      p[12 + i] = p_edge(a3[i], z3[i], d3[i], g3[i], s1r, bcv, eM);
    }
#pragma unroll
    for (int i = 0; i < 16; ++i) lsum += p[i];

    union { v16bf v; unsigned u[8]; } A;
#pragma unroll
    for (int d = 0; d < 8; ++d) A.u[d] = pack2_bf16(p[2 * d], p[2 * d + 1]);

    // 8 output-dim tiles: B = hbT[n][k0 + hi*8 .. +7] and hbT[n][k0 + 16 + hi*8 .. +7]
#pragma unroll
    for (int t = 0; t < 8; ++t) {
      const ushort_t* bp = hbT + (size_t)(t * 16 + r) * NN + k0 + hi * 8;
      union { v16bf v; v4u q[2]; } B;
      B.q[0] = *(const v4u*)bp;
      B.q[1] = *(const v4u*)(bp + 16);
      c[t] = __builtin_amdgcn_wmma_f32_16x16x32_bf16(
          false, A.v, false, B.v, (short)0, c[t], false, false);
    }
  }

  // combine 8 waves' partials in LDS
  atomicAdd(&l_lds[r], lsum);
#pragma unroll
  for (int t = 0; t < 8; ++t)
#pragma unroll
    for (int v = 0; v < 8; ++v)
      atomicAdd(&o_lds[(v + 8 * hi) * DD + t * 16 + r], c[t][v]);
  __syncthreads();

  for (int i = tid; i < 16 * DD; i += 256) {
    const int rr = i >> 7, cc = i & 127;
    out[(size_t)(row0 + rr) * DD + cc] = o_lds[i] / (l_lds[rr] + 1e-9f);
  }
}

// ---------------------------------------------------------------------------
extern "C" void kernel_launch(void* const* d_in, const int* in_sizes, int n_in,
                              void* d_out, int out_size, void* d_ws, size_t ws_size,
                              hipStream_t stream) {
  const float* x      = (const float*)d_in[0];
  const int*   adj    = (const int*)d_in[1];
  const float* dist   = (const float*)d_in[2];
  const float* ang    = (const float*)d_in[3];
  const float* W      = (const float*)d_in[4];
  const float* attn_w = (const float*)d_in[5];
  const float* attn_b = (const float*)d_in[6];
  float* out = (float*)d_out;

  char* ws = (char*)d_ws;
  ushort_t* hbT = (ushort_t*)ws;                          // DD*NN*2 = 1 MB
  float* s1  = (float*)(ws + (size_t)DD * NN * 2);        // 16 KB
  float* s2  = s1 + NN;                                   // 16 KB
  float* s2m = s2 + NN;                                   // 4 B

  gat_h_kernel<<<NN / 32, 256, 0, stream>>>(x, W, attn_w, hbT, s1, s2);
  gat_s2max_kernel<<<1, 256, 0, stream>>>(s2, s2m);
  gat_attn_kernel<<<NN / 16, 256, 0, stream>>>(adj, dist, ang, hbT, s1, s2, s2m,
                                               attn_b, out);
}
